// COTAttention_89034672046291
// MI455X (gfx1250) — matrix operand order repaired and checked by reading, not compiled, not run on hardware
//
#include <hip/hip_runtime.h>

// ---------------------------------------------------------------------------
// COTAttention forward for MI455X (gfx1250, wave32, WMMA bf16 16x16x32).
// GEMM-like convs -> v_wmma_f32_16x16x32_bf16, A-tiles via async LDS copies
// (ASYNCcnt) when the toolchain exposes the builtin, branchless im2col gather.
// ---------------------------------------------------------------------------

typedef __attribute__((ext_vector_type(16))) __bf16 v16bf;
typedef __attribute__((ext_vector_type(8)))  __bf16 v8bf;
typedef __attribute__((ext_vector_type(8)))  float  v8f;
typedef int i128v __attribute__((vector_size(16)));   // 16B chunk for async copy

#define SZ   33554432LL   // B*C*H*W = 2*256*256*256
#define HW   65536
#define Cd   256

#define GAS __attribute__((address_space(1)))
#define LAS __attribute__((address_space(3)))

#ifndef __has_builtin
#define __has_builtin(x) 0
#endif
#if __has_builtin(__builtin_amdgcn_global_load_async_to_lds_b128)
#define HAVE_ASYNC_LDS 1
#else
#define HAVE_ASYNC_LDS 0
#endif

union Frag  { v16bf v; v8bf h[2]; };
union BVec  { v8bf v; unsigned short u[8]; };

__device__ __forceinline__ unsigned short f2bf(float x) {
  unsigned u = __float_as_uint(x);
  unsigned r = u + 0x7FFFu + ((u >> 16) & 1u);
  return (unsigned short)(r >> 16);
}
__device__ __forceinline__ float bf2f(unsigned short h) {
  return __uint_as_float(((unsigned)h) << 16);
}

// copy one 16-byte chunk global->LDS (async if available)
__device__ __forceinline__ void cp16(const unsigned short* g, unsigned short* l) {
#if HAVE_ASYNC_LDS
  __builtin_amdgcn_global_load_async_to_lds_b128((GAS i128v*)g, (LAS i128v*)l, 0, 0);
#else
  *(v8bf*)l = *(const v8bf*)g;
#endif
}
__device__ __forceinline__ void wait_async0() {
#if HAVE_ASYNC_LDS
#if __has_builtin(__builtin_amdgcn_s_wait_asynccnt)
  __builtin_amdgcn_s_wait_asynccnt(0);
#else
  asm volatile("s_wait_asynccnt 0x0" ::: "memory");
#endif
#endif
}

// One 16x16 (MxN) x K=32 WMMA step. As: [rows][LDW] bf16 (M-major tile),
// Bs: [cols][LDW] bf16 (B transposed: N-major, K contiguous), koff = 0/32.
// A frag (ISA 7.12.2): lane l -> row wm+(l&15), K runs {kb..kb+7, kb+16..kb+23}.
// B frag: lane l -> col wn+(l&15), K runs {kb..kb+7, kb+16..kb+23} (same as A).
template <int LDW>
__device__ __forceinline__ v8f wmma_t(const unsigned short* As,
                                      const unsigned short* Bs,
                                      int wm, int wn, int lane, v8f acc, int koff) {
  Frag a, b;
  int kb  = koff + (lane >> 4) * 8;
  int r   = wm + (lane & 15);
  a.h[0] = *(const v8bf*)(As + r * LDW + kb);
  a.h[1] = *(const v8bf*)(As + r * LDW + kb + 16);
  int c   = wn + (lane & 15);
  int kb2 = koff + (lane >> 4) * 8;
  b.h[0] = *(const v8bf*)(Bs + c * LDW + kb2);
  b.h[1] = *(const v8bf*)(Bs + c * LDW + kb2 + 16);
  return __builtin_amdgcn_wmma_f32_16x16x32_bf16(false, a.v, false, b.v,
                                                 (short)0, acc, false, false);
}

struct Epi {
  float* Cf;                 // fp32 dest (or null)
  unsigned short* Cb;        // bf16 dest (or null)
  long long ONPB, OBS, ORS;  // dest = (n/ONPB)*OBS + row*ORS + (n%ONPB)
  const float* scale;
  const float* bias;
  int relu6;
};

// C frag layout: lane l holds acc[j] = C[mbase + 8*(l>>4) + j][nbase + (l&15)]
__device__ __forceinline__ void epi_store(const Epi& e, v8f acc,
                                          int mbase, long long nbase, int lane) {
  long long n   = nbase + (lane & 15);
  long long img = n / e.ONPB, pp = n % e.ONPB;
  int rb = mbase + ((lane >> 4) << 3);
#pragma unroll
  for (int j = 0; j < 8; ++j) {
    int row = rb + j;
    float v = acc[j];
    if (e.scale) v *= e.scale[row];
    if (e.bias)  v += e.bias[row];
    if (e.relu6) v = fminf(fmaxf(v, 0.f), 6.f);
    long long d = img * e.OBS + (long long)row * e.ORS + pp;
    if (e.Cf) e.Cf[d] = v;
    else      e.Cb[d] = f2bf(v);
  }
}

// 64x64 A-tile load (K-stage 64): 512 16B chunks, 2 per thread
__device__ __forceinline__ void load_A64(const unsigned short* Ag, unsigned short* As,
                                         long long rowstride, int tid) {
#pragma unroll
  for (int h = 0; h < 2; ++h) {
    int c2 = tid * 2 + h;
    int r = c2 >> 3, cc = (c2 & 7) * 8;
    cp16(Ag + (long long)r * rowstride + cc, As + r * 64 + cc);
  }
}

// ------------------ generic 1x1 GEMM, BM=64, K%64==0 (WMMA) ----------------
struct GemmP {
  const unsigned short* A;   // [M][K] bf16 row-major
  const unsigned short* B;   // input bf16, batched channel-major
  int K;
  long long NPB, BSTRIDE;    // n per slab, elements per slab
  long long KROW;            // k row stride inside slab
  Epi e;
};

__global__ __launch_bounds__(256) void k_gemm64(GemmP p) {
  __shared__ __align__(32) unsigned short As[64 * 64];
  __shared__ __align__(32) unsigned short Bs[64 * 64];
  int tid = threadIdx.x, lane = tid & 31, wave = tid >> 5;
  long long n0 = (long long)blockIdx.x * 64;
  int m0 = blockIdx.y * 64;
  int wm = (wave >> 1) * 16, wn = (wave & 1) * 32;
  v8f acc0 = {}; v8f acc1 = {};
  long long img = n0 / p.NPB;
  const unsigned short* Bb = p.B + img * p.BSTRIDE + (n0 % p.NPB);
  for (int k0 = 0; k0 < p.K; k0 += 64) {
    load_A64(p.A + (long long)m0 * p.K + k0, As, p.K, tid);
    {
      int kk = tid >> 2, nc = (tid & 3) * 16;
      const unsigned short* g = Bb + (long long)(k0 + kk) * p.KROW + nc;
      BVec t0, t1;
      t0.v = *(const v8bf*)g;
      t1.v = *(const v8bf*)(g + 8);
#pragma unroll
      for (int j = 0; j < 8; ++j) {
        Bs[(nc + j) * 64 + kk]     = t0.u[j];
        Bs[(nc + 8 + j) * 64 + kk] = t1.u[j];
      }
    }
    if (k0 + 64 < p.K)
      __builtin_prefetch(p.A + (long long)(m0 + (tid >> 2)) * p.K + k0 + 64, 0, 0);
    wait_async0();
    __syncthreads();
#pragma unroll
    for (int q = 0; q < 2; ++q) {
      acc0 = wmma_t<64>(As, Bs, wm, wn,      lane, acc0, q * 32);
      acc1 = wmma_t<64>(As, Bs, wm, wn + 16, lane, acc1, q * 32);
    }
    __syncthreads();
  }
  epi_store(p.e, acc0, m0 + wm, n0 + wn, lane);
  epi_store(p.e, acc1, m0 + wm, n0 + wn + 16, lane);
}

// ------------------ small GEMM, BM=32, K-step 32 (vemb) --------------------
__global__ __launch_bounds__(256) void k_gemm32(GemmP p) {
  __shared__ __align__(32) unsigned short As[32 * 32];
  __shared__ __align__(32) unsigned short Bs[64 * 32];
  int tid = threadIdx.x, lane = tid & 31, wave = tid >> 5;
  long long n0 = (long long)blockIdx.x * 64;
  int wm = (wave >> 2) * 16, wn = (wave & 3) * 16;
  v8f acc = {};
  long long img = n0 / p.NPB;
  const unsigned short* Bb = p.B + img * p.BSTRIDE + (n0 % p.NPB);
  for (int k0 = 0; k0 < p.K; k0 += 32) {
    if (tid < 128) {
      int r = tid >> 2, cc = (tid & 3) * 8;
      cp16(p.A + (long long)r * p.K + k0 + cc, As + r * 32 + cc);
    }
    {
      int kk = tid >> 3, nc = (tid & 7) * 8;
      BVec t;
      t.v = *(const v8bf*)(Bb + (long long)(k0 + kk) * p.KROW + nc);
#pragma unroll
      for (int j = 0; j < 8; ++j) Bs[(nc + j) * 32 + kk] = t.u[j];
    }
    wait_async0();
    __syncthreads();
    acc = wmma_t<32>(As, Bs, wm, wn, lane, acc, 0);
    __syncthreads();
  }
  epi_store(p.e, acc, wm, n0 + wn, lane);
}

// ---------------- implicit-GEMM conv3x3 over BCHW (local path) -------------
struct ConvLP { const unsigned short* A; const unsigned short* X; Epi e; };

__global__ __launch_bounds__(256) void k_conv_local(ConvLP p) {
  // M=256 (BM=64), K=2304=256*9 (36 stages of 64), N tiled 64 inside one row
  __shared__ __align__(32) unsigned short As[64 * 64];
  __shared__ __align__(32) unsigned short Bs[64 * 64];
  int tid = threadIdx.x, lane = tid & 31, wave = tid >> 5;
  long long n0 = (long long)blockIdx.x * 64;
  int m0 = blockIdx.y * 64;
  int b  = (int)(n0 >> 16);
  int y  = ((int)(n0 >> 8)) & 255;
  int x0 = (int)n0 & 255;
  int wm = (wave >> 1) * 16, wn = (wave & 1) * 32;
  v8f acc0 = {}; v8f acc1 = {};
  for (int k0 = 0; k0 < 2304; k0 += 64) {
    load_A64(p.A + (long long)m0 * 2304 + k0, As, 2304, tid);
    {
      int kk = tid >> 2, nc = (tid & 3) * 16;
      int k = k0 + kk, ic = k / 9, tap = k - ic * 9;
      int dy = tap / 3 - 1, dx = tap - (tap / 3) * 3 - 1;
      int ys = y + dy;
      int ysc = min(max(ys, 0), 255);
      bool yok = (unsigned)ys < 256u;
      const unsigned short* src = p.X + (((long long)b * Cd + ic) * 256 + ysc) * 256;
#pragma unroll
      for (int j = 0; j < 16; ++j) {
        int xs = x0 + nc + j + dx;
        int xsc = min(max(xs, 0), 255);
        unsigned short v = src[xsc];                 // always load (clamped)
        if (!(yok && (unsigned)xs < 256u)) v = 0;    // v_cndmask, no exec branch
        Bs[(nc + j) * 64 + kk] = v;
      }
    }
    wait_async0();
    __syncthreads();
#pragma unroll
    for (int q = 0; q < 2; ++q) {
      acc0 = wmma_t<64>(As, Bs, wm, wn,      lane, acc0, q * 32);
      acc1 = wmma_t<64>(As, Bs, wm, wn + 16, lane, acc1, q * 32);
    }
    __syncthreads();
  }
  epi_store(p.e, acc0, m0 + wm, n0 + wn, lane);
  epi_store(p.e, acc1, m0 + wm, n0 + wn + 16, lane);
}

// ---------- implicit-GEMM conv3x3 over window tensors (kemb path) ----------
__global__ __launch_bounds__(256) void k_conv_win(ConvLP p) {
  // input [2048 imgs][32 ch][64 x 8], M=32 (BM=32), K=288=9*32
  __shared__ __align__(32) unsigned short As[32 * 32];
  __shared__ __align__(32) unsigned short Bs[64 * 32];
  int tid = threadIdx.x, lane = tid & 31, wave = tid >> 5;
  long long n0 = (long long)blockIdx.x * 64;
  int img = (int)(n0 >> 9);
  int p0  = (int)n0 & 511;
  int y0  = p0 >> 3;
  int wm = (wave >> 2) * 16, wn = (wave & 3) * 16;
  v8f acc = {};
  const unsigned short* src0 = p.X + (long long)img * 16384;
  for (int k0 = 0; k0 < 288; k0 += 32) {
    if (tid < 128) {
      int r = tid >> 2, cc = (tid & 3) * 8;
      cp16(p.A + (long long)r * 288 + k0 + cc, As + r * 32 + cc);
    }
    {
      int kk = tid >> 3, nc = (tid & 7) * 8;
      int k = k0 + kk, ic = k / 9, tap = k - ic * 9;
      int dy = tap / 3 - 1, dx = tap - (tap / 3) * 3 - 1;
#pragma unroll
      for (int j = 0; j < 8; ++j) {
        int e = nc + j;
        int yy = y0 + (e >> 3) + dy;
        int xx = (e & 7) + dx;
        int yyc = min(max(yy, 0), 63), xxc = min(max(xx, 0), 7);
        unsigned short v = src0[ic * 512 + yyc * 8 + xxc];
        if (!((unsigned)yy < 64u && (unsigned)xx < 8u)) v = 0;
        Bs[e * 32 + kk] = v;
      }
    }
    wait_async0();
    __syncthreads();
    acc = wmma_t<32>(As, Bs, wm, wn, lane, acc, 0);
    __syncthreads();
  }
  epi_store(p.e, acc, wm, n0 + wn, lane);
}

// ------------- qkv 1x1 GEMM with fused window-rearrange epilogue -----------
struct QkvP {
  const unsigned short* A;   // [768][256] bf16
  const unsigned short* X;   // x bf16 BCHW
  unsigned short *Q, *Kd, *V;
};

__device__ __forceinline__ void qkv_store(const QkvP& p, v8f acc,
                                          int mbase, long long nbase, int lane) {
  long long n = nbase + (lane & 15);
  int b  = (int)(n >> 16);
  int pp = (int)n & 65535;
  int y = pp >> 8, x = pp & 255;
  int hh = y >> 3, ws1 = y & 7, ww = x >> 3, ws2 = x & 7;
  int rb = mbase + ((lane >> 4) << 3);
#pragma unroll
  for (int j = 0; j < 8; ++j) {
    int row = rb + j;
    int t3 = row >> 8, c = row & 255;
    int head = c >> 3, dd = c & 7;
    long long d = ((((long long)(b * 32 + hh) * 32 + ww) * 32 + head) * 512) +
                  (ws1 * 8 + ws2) * 8 + dd;
    unsigned short v = f2bf(acc[j]);
    if (t3 == 0) p.Q[d] = v; else if (t3 == 1) p.Kd[d] = v; else p.V[d] = v;
  }
}

__global__ __launch_bounds__(256) void k_qkv(QkvP p) {
  __shared__ __align__(32) unsigned short As[64 * 64];
  __shared__ __align__(32) unsigned short Bs[64 * 64];
  int tid = threadIdx.x, lane = tid & 31, wave = tid >> 5;
  long long n0 = (long long)blockIdx.x * 64;
  int m0 = blockIdx.y * 64;
  int wm = (wave >> 1) * 16, wn = (wave & 1) * 32;
  v8f acc0 = {}; v8f acc1 = {};
  long long img = n0 >> 16;
  const unsigned short* Bb = p.X + img * ((long long)Cd * HW) + (n0 & 65535);
  for (int k0 = 0; k0 < 256; k0 += 64) {
    load_A64(p.A + (long long)m0 * 256 + k0, As, 256, tid);
    {
      int kk = tid >> 2, nc = (tid & 3) * 16;
      const unsigned short* g = Bb + (long long)(k0 + kk) * HW + nc;
      BVec t0, t1;
      t0.v = *(const v8bf*)g;
      t1.v = *(const v8bf*)(g + 8);
#pragma unroll
      for (int j = 0; j < 8; ++j) {
        Bs[(nc + j) * 64 + kk]     = t0.u[j];
        Bs[(nc + 8 + j) * 64 + kk] = t1.u[j];
      }
    }
    wait_async0();
    __syncthreads();
#pragma unroll
    for (int q = 0; q < 2; ++q) {
      acc0 = wmma_t<64>(As, Bs, wm, wn,      lane, acc0, q * 32);
      acc1 = wmma_t<64>(As, Bs, wm, wn + 16, lane, acc1, q * 32);
    }
    __syncthreads();
  }
  qkv_store(p, acc0, m0 + wm, n0 + wn, lane);
  qkv_store(p, acc1, m0 + wm, n0 + wn + 16, lane);
}

// ----------------------------- prep kernels --------------------------------
__global__ void k_cvt_bf(const float* in, unsigned short* out, long long n) {
  long long i = (long long)blockIdx.x * 256 + threadIdx.x;
  if (i < n) out[i] = f2bf(in[i]);
}

__global__ void k_fold_local(const float* w1, const float* w2,
                             const float* g1, const float* b1, const float* m1, const float* v1,
                             const float* g2, const float* b2, const float* m2, const float* v2,
                             unsigned short* wc, float* lb) {
  int i = blockIdx.x * 256 + threadIdx.x;
  if (i >= 256 * 2304) return;
  int oc = i / 2304, k = i - oc * 2304, ic = k / 9, tap = k - ic * 9;
  float s1 = g1[oc] * rsqrtf(v1[oc] + 1e-5f);
  float val = w1[i] * s1;
  if (tap == 4) {
    float s2 = g2[oc] * rsqrtf(v2[oc] + 1e-5f);
    val += w2[oc * 256 + ic] * s2;
  }
  wc[i] = f2bf(val);
  if (k == 0) {
    float s2 = g2[oc] * rsqrtf(v2[oc] + 1e-5f);
    lb[oc] = (b1[oc] - m1[oc] * s1) + (b2[oc] - m2[oc] * s2);
  }
}

__global__ void k_fold_cw(const float* w, const float* g, const float* b,
                          const float* m, const float* v,
                          int ksz, int total, unsigned short* wo, float* bias) {
  int i = blockIdx.x * 256 + threadIdx.x;
  if (i >= total) return;
  int oc = i / ksz;
  float s = g[oc] * rsqrtf(v[oc] + 1e-5f);
  wo[i] = f2bf(w[i] * s);
  if (i - oc * ksz == 0) bias[oc] = b[oc] - m[oc] * s;
}

__global__ void k_fold_attdw(const float* wdw, const float* g, const float* b,
                             const float* m, const float* v, float* sc, float* tb) {
  int i = threadIdx.x;
  if (i >= 64) return;
  float s = g[i] * rsqrtf(v[i] + 1e-5f);
  sc[i] = wdw[i] * s;
  tb[i] = b[i] - m[i] * s;
}

__global__ void k_fold_bn(const float* g, const float* b, const float* m,
                          const float* v, int n, float* s, float* t) {
  int i = blockIdx.x * 256 + threadIdx.x;
  if (i >= n) return;
  float ss = g[i] * rsqrtf(v[i] + 1e-5f);
  s[i] = ss; t[i] = b[i] - m[i] * ss;
}

// a = relu6(bn(dw1x1(concat([k1, q]))))  -> bf16 [2048][64][512]
__global__ void k_make_a(const float* k1, const unsigned short* q,
                         const float* sc, const float* tb, unsigned short* a) {
  long long i = (long long)blockIdx.x * 256 + threadIdx.x;
  if (i >= 2 * SZ) return;
  int pp = (int)i & 511;
  int ch = (int)((i >> 9) & 63);
  long long img = i >> 15;
  float val = (ch < 32) ? k1[img * 16384 + (long long)ch * 512 + pp]
                        : bf2f(q[img * 16384 + (long long)(ch - 32) * 512 + pp]);
  float r = val * sc[ch] + tb[ch];
  r = fminf(fmaxf(r, 0.f), 6.f);
  a[i] = f2bf(r);
}

// mean over dim 2 of flat reshape(B,C,2,H,W)
__global__ void k_mean(const float* attn, float* am) {
  long long i = (long long)blockIdx.x * 256 + threadIdx.x;
  if (i >= SZ) return;
  long long bc = i >> 16;
  int pp = (int)i & 65535;
  long long f = (bc << 17) + pp;
  am[i] = 0.5f * (attn[f] + attn[f + HW]);
}

// softmax over HW per (b,c) row, fused k2 = softmax * vv
__global__ __launch_bounds__(1024) void k_softmax_mul(const float* am, const float* vv, float* k2) {
  __shared__ float red[32];
  int row = blockIdx.x, tid = threadIdx.x;
  const float* xr = am + (long long)row * HW;
  const float* vr = vv + (long long)row * HW;
  float* orow = k2 + (long long)row * HW;
  float mx = -3.4e38f;
  for (int i = tid; i < HW; i += 1024) mx = fmaxf(mx, xr[i]);
  for (int o = 16; o; o >>= 1) mx = fmaxf(mx, __shfl_xor(mx, o, 32));
  if ((tid & 31) == 0) red[tid >> 5] = mx;
  __syncthreads();
  if (tid < 32) {
    float t = red[tid];
    for (int o = 16; o; o >>= 1) t = fmaxf(t, __shfl_xor(t, o, 32));
    red[tid] = t;
  }
  __syncthreads();
  mx = red[0];
  __syncthreads();
  float sm = 0.f;
  for (int i = tid; i < HW; i += 1024) sm += __expf(xr[i] - mx);
  for (int o = 16; o; o >>= 1) sm += __shfl_xor(sm, o, 32);
  if ((tid & 31) == 0) red[tid >> 5] = sm;
  __syncthreads();
  if (tid < 32) {
    float t = red[tid];
    for (int o = 16; o; o >>= 1) t += __shfl_xor(t, o, 32);
    red[tid] = t;
  }
  __syncthreads();
  float inv = 1.f / red[0];
  for (int i = tid; i < HW; i += 1024) orow[i] = __expf(xr[i] - mx) * inv * vr[i];
}

// ox + oy (reflect-pad avgpools) + k1(flat) + local
__global__ void k_poolsum(const float* k2, const float* k1, const float* local, float* outp) {
  long long i = (long long)blockIdx.x * 256 + threadIdx.x;
  if (i >= SZ) return;
  int w = (int)i & 255, h = ((int)(i >> 8)) & 255;
  long long bc = i >> 16;
  const float* base = k2 + (bc << 16);
  float sx = 0.f, sy = 0.f;
#pragma unroll
  for (int t = 0; t < 8; ++t) {
    int hh = h - 3 + t;
    if (hh >= 0 && hh <= 256) { int r = (hh == 256) ? 254 : hh; sx += base[r * 256 + w]; }
    int wi = w - 3 + t;
    if (wi >= 0 && wi <= 256) { int c = (wi == 256) ? 254 : wi; sy += base[h * 256 + c]; }
  }
  outp[i] = 0.125f * (sx + sy) + k1[i] + local[i];
}

// depthwise 8x8 proj conv (reflect-pad +1 then zero-pad 3), fused BN -> bf16
__global__ __launch_bounds__(256) void k_dwconv(const float* inp, const float* wdw,
                                                const float* s, const float* t,
                                                unsigned short* out) {
  __shared__ float wsh[64];
  int bc = blockIdx.y;
  int c = bc & 255;
  int pp = blockIdx.x * 256 + threadIdx.x;
  if (threadIdx.x < 64) wsh[threadIdx.x] = wdw[c * 64 + threadIdx.x];
  __syncthreads();
  int y = pp >> 8, x = pp & 255;
  const float* base = inp + ((long long)bc << 16);
  float acc = 0.f;
#pragma unroll
  for (int ii = 0; ii < 8; ++ii) {
    int m = y + ii - 3;
    if (m < 0 || m > 256) continue;
    int mm = (m == 256) ? 254 : m;
    const float* rowp = base + mm * 256;
#pragma unroll
    for (int jj = 0; jj < 8; ++jj) {
      int n = x + jj - 3;
      if (n < 0 || n > 256) continue;
      int nc = (n == 256) ? 254 : n;
      acc += wsh[ii * 8 + jj] * rowp[nc];
    }
  }
  out[((long long)bc << 16) + pp] = f2bf(acc * s[c] + t[c]);
}

// ---------------------------------------------------------------------------
extern "C" void kernel_launch(void* const* d_in, const int* in_sizes, int n_in,
                              void* d_out, int out_size, void* d_ws, size_t ws_size,
                              hipStream_t stream) {
  (void)in_sizes; (void)n_in; (void)out_size; (void)ws_size;
  const float* x        = (const float*)d_in[0];
  const float* qkv_w    = (const float*)d_in[1];
  const float* local1_w = (const float*)d_in[2];
  const float* local2_w = (const float*)d_in[3];
  const float* kemb_w   = (const float*)d_in[4];
  const float* vemb_w   = (const float*)d_in[5];
  const float* att_dw_w = (const float*)d_in[6];
  const float* att_pw_w = (const float*)d_in[7];
  const float* proj_dw_w= (const float*)d_in[8];
  const float* proj_pw_w= (const float*)d_in[9];
  const float* l1g = (const float*)d_in[10]; const float* l1b = (const float*)d_in[11];
  const float* l2g = (const float*)d_in[12]; const float* l2b = (const float*)d_in[13];
  const float* keg = (const float*)d_in[14]; const float* keb = (const float*)d_in[15];
  const float* veg = (const float*)d_in[16]; const float* veb = (const float*)d_in[17];
  const float* atg = (const float*)d_in[18]; const float* atb = (const float*)d_in[19];
  const float* prg = (const float*)d_in[20]; const float* prb = (const float*)d_in[21];
  const float* l1m = (const float*)d_in[22]; const float* l1v = (const float*)d_in[23];
  const float* l2m = (const float*)d_in[24]; const float* l2v = (const float*)d_in[25];
  const float* kem = (const float*)d_in[26]; const float* kev = (const float*)d_in[27];
  const float* vem = (const float*)d_in[28]; const float* vev = (const float*)d_in[29];
  const float* atm = (const float*)d_in[30]; const float* atv = (const float*)d_in[31];
  const float* prm = (const float*)d_in[32]; const float* prv = (const float*)d_in[33];

  const size_t MiB = 1ull << 20;
  char* base = (char*)d_ws;
  // ---- weight / small buffers (first 4 MiB) ----
  unsigned short* WC  = (unsigned short*)(base + 0);            // 256x2304 bf16
  float*  LB   = (float*) (base + 1536 * 1024);
  unsigned short* QW  = (unsigned short*)(base + 1600 * 1024);  // 768x256
  unsigned short* KEW = (unsigned short*)(base + 2048 * 1024);  // 32x288
  float*  KEB  = (float*) (base + 2112 * 1024);
  unsigned short* VEW = (unsigned short*)(base + 2176 * 1024);  // 32x32
  float*  VEB  = (float*) (base + 2240 * 1024);
  unsigned short* ATW = (unsigned short*)(base + 2304 * 1024);  // 64x64
  float*  ADWs = (float*) (base + 2368 * 1024);
  float*  ADWt = (float*) (base + 2432 * 1024);
  float*  PS   = (float*) (base + 2496 * 1024);
  float*  PT   = (float*) (base + 2560 * 1024);
  unsigned short* PW  = (unsigned short*)(base + 2624 * 1024);  // 256x256
  // ---- big buffers (time-sliced reuse plan, peak 900 MiB) ----
  unsigned short* XBF  = (unsigned short*)(base +   4 * MiB);   // 64 MiB
  float*  LOCAL = (float*) (base +  68 * MiB);                  // 128
  float*  K1    = (float*) (base + 196 * MiB);                  // 128
  float*  VV    = (float*) (base + 324 * MiB);                  // 128
  unsigned short* QBF = (unsigned short*)(base + 452 * MiB);    // 64
  unsigned short* KBF = (unsigned short*)(base + 516 * MiB);    // 64
  unsigned short* VBF = (unsigned short*)(base + 580 * MiB);    // 64
  unsigned short* ABF = (unsigned short*)(base + 516 * MiB);    // 128 (over KBF+VBF)
  float*  ATTN  = (float*) (base + 644 * MiB);                  // 256
  float*  AM    = (float*) (base + 452 * MiB);                  // 128 (over QBF+ABF)
  float*  K2    = (float*) (base + 644 * MiB);                  // 128 (over ATTN)
  float*  OUTP  = (float*) (base + 324 * MiB);                  // 128 (over VV)
  unsigned short* DWBF = (unsigned short*)(base + 4 * MiB);     // 64 (over XBF)

  // ---- weight prep ----
  k_fold_local<<<2304, 256, 0, stream>>>(local1_w, local2_w,
      l1g, l1b, l1m, l1v, l2g, l2b, l2m, l2v, WC, LB);
  k_cvt_bf<<<768, 256, 0, stream>>>(qkv_w, QW, 196608);
  k_cvt_bf<<<16, 256, 0, stream>>>(att_pw_w, ATW, 4096);
  k_cvt_bf<<<256, 256, 0, stream>>>(proj_pw_w, PW, 65536);
  k_fold_cw<<<36, 256, 0, stream>>>(kemb_w, keg, keb, kem, kev, 288, 9216, KEW, KEB);
  k_fold_cw<<<4, 256, 0, stream>>>(vemb_w, veg, veb, vem, vev, 32, 1024, VEW, VEB);
  k_fold_attdw<<<1, 64, 0, stream>>>(att_dw_w, atg, atb, atm, atv, ADWs, ADWt);
  k_fold_bn<<<1, 256, 0, stream>>>(prg, prb, prm, prv, 256, PS, PT);

  // ---- x -> bf16 ----
  k_cvt_bf<<<131072, 256, 0, stream>>>(x, XBF, SZ);

  // ---- local = folded conv3x3 (implicit GEMM, WMMA) ----
  {
    ConvLP p; p.A = WC; p.X = XBF;
    p.e = Epi{LOCAL, nullptr, 65536, (long long)Cd * HW, HW, nullptr, LB, 0};
    k_conv_local<<<dim3(2048, 4), 256, 0, stream>>>(p);
  }
  // ---- qkv 1x1 GEMM with window rearrange ----
  {
    QkvP p; p.A = QW; p.X = XBF; p.Q = QBF; p.Kd = KBF; p.V = VBF;
    k_qkv<<<dim3(2048, 12), 256, 0, stream>>>(p);
  }
  // ---- kemb conv3x3 on windows -> k1 (relu6(bn)) ----
  {
    ConvLP p; p.A = KEW; p.X = KBF;
    p.e = Epi{K1, nullptr, 512, 16384, 512, nullptr, KEB, 1};
    k_conv_win<<<dim3(16384, 1), 256, 0, stream>>>(p);
  }
  // ---- vemb 1x1 -> vv ----
  {
    GemmP p; p.A = VEW; p.B = VBF; p.K = 32;
    p.NPB = 512; p.BSTRIDE = 16384; p.KROW = 512;
    p.e = Epi{VV, nullptr, 512, 16384, 512, nullptr, VEB, 0};
    k_gemm32<<<dim3(16384, 1), 256, 0, stream>>>(p);
  }
  // ---- a = relu6(bn(dw(concat(k1,q)))) -> bf16 ----
  k_make_a<<<262144, 256, 0, stream>>>(K1, QBF, ADWs, ADWt, ABF);
  // ---- attn = att pointwise GEMM (K=64, one 64-stage) ----
  {
    GemmP p; p.A = ATW; p.B = ABF; p.K = 64;
    p.NPB = 512; p.BSTRIDE = 32768; p.KROW = 512;
    p.e = Epi{ATTN, nullptr, 512, 32768, 512, nullptr, nullptr, 0};
    k_gemm64<<<dim3(16384, 1), 256, 0, stream>>>(p);
  }
  // ---- mean -> softmax(*vv) -> k2 ----
  k_mean<<<131072, 256, 0, stream>>>(ATTN, AM);
  k_softmax_mul<<<512, 1024, 0, stream>>>(AM, VV, K2);
  // ---- pools + sums ----
  k_poolsum<<<131072, 256, 0, stream>>>(K2, K1, LOCAL, OUTP);
  // ---- proj: depthwise 8x8 + BN -> bf16, then pointwise GEMM -> d_out ----
  k_dwconv<<<dim3(256, 512), 256, 0, stream>>>(OUTP, proj_dw_w, PS, PT, DWBF);
  {
    GemmP p; p.A = PW; p.B = DWBF; p.K = 256;
    p.NPB = 65536; p.BSTRIDE = (long long)Cd * HW; p.KROW = 65536;
    p.e = Epi{(float*)d_out, nullptr, 65536, (long long)Cd * HW, HW, nullptr, nullptr, 0};
    k_gemm64<<<dim3(2048, 4), 256, 0, stream>>>(p);
  }
}
